// TopKRouter_55362128446066
// MI455X (gfx1250) — matrix operand order repaired and checked by reading, not compiled, not run on hardware
//
#include <hip/hip_runtime.h>

typedef __attribute__((ext_vector_type(16))) _Float16 v16h;
typedef __attribute__((ext_vector_type(8)))  _Float16 v8h;
typedef __attribute__((ext_vector_type(4)))  _Float16 v4h;
typedef __attribute__((ext_vector_type(8)))  float    v8f;
typedef __attribute__((ext_vector_type(4)))  float    v4f;

#define DDIM   4096
#define NTOK   16384      // B*T = 4*4096
#define NEXP   8
#define KSEL   2
#define WAVES  4
#define BLOCK  (WAVES * 32)
#define TOK_PER_WG (WAVES * 16)

// LDS: 8 expert rows x 4096 f16 = 65536 bytes (rotated per row to spread banks).
// The per-token logit scratch (2 KB) is overlaid on the same space after a barrier.
#define SMEM_BYTES 65536

__global__ __launch_bounds__(BLOCK)
void topk_router_wmma(const float* __restrict__ x,
                      const float* __restrict__ Wg,
                      float* __restrict__ wout,
                      int*   __restrict__ iout)
{
    extern __shared__ char smem[];
    _Float16* Wh = (_Float16*)smem;   // [8][4096] halfs, row e rotated by e*8 halfs

    const int tid = threadIdx.x;

    // ---- Phase 0: W (8 x 4096 f32, L2-resident) -> LDS f16, rotated rows ----
    for (int idx = tid; idx < NEXP * (DDIM / 4); idx += BLOCK) {
        const int e  = idx >> 10;            // DDIM/4 = 1024 quads per row
        const int dq = (idx & 1023) << 2;    // half index, multiple of 4
        const v4f f = *(const v4f*)(Wg + e * DDIM + dq);
        v4h h;
        h[0] = (_Float16)f[0]; h[1] = (_Float16)f[1];
        h[2] = (_Float16)f[2]; h[3] = (_Float16)f[3];
        const int col = (dq + e * 8) & (DDIM - 1);   // rotate row e by e*16B
        *(v4h*)&Wh[e * DDIM + col] = h;
    }
    __syncthreads();

    // ---- Phase 1: GEMM via v_wmma_f32_16x16x32_f16 ----
    const int lane   = tid & 31;
    const int wave   = tid >> 5;
    const int lanelo = lane & 15;     // A: token row; B: expert column
    const int lanehi = lane >> 4;     // K-half selector
    const int exp8   = lanelo & 7;    // experts 8..15 duplicate 0..7 (ignored later)

    const int tok0 = blockIdx.x * TOK_PER_WG + wave * 16;
    const float* px = x + (size_t)(tok0 + lanelo) * DDIM + lanehi * 8;

    v8f acc = {};
    #pragma unroll 2
    for (int c = 0; c < DDIM / 32; ++c) {
        const int d0 = c * 32;
        // A: this lane needs d0+[hi*8,+8) and d0+16+[hi*8,+8) (one 128B line
        // per token row per chunk across the lo/hi lane pair).
        const v4f f0 = *(const v4f*)(px + d0);
        const v4f f1 = *(const v4f*)(px + d0 + 4);
        const v4f f2 = *(const v4f*)(px + d0 + 16);
        const v4f f3 = *(const v4f*)(px + d0 + 20);
        v16h a;
        a[0]=(_Float16)f0[0];  a[1]=(_Float16)f0[1];  a[2]=(_Float16)f0[2];  a[3]=(_Float16)f0[3];
        a[4]=(_Float16)f1[0];  a[5]=(_Float16)f1[1];  a[6]=(_Float16)f1[2];  a[7]=(_Float16)f1[3];
        a[8]=(_Float16)f2[0];  a[9]=(_Float16)f2[1];  a[10]=(_Float16)f2[2]; a[11]=(_Float16)f2[3];
        a[12]=(_Float16)f3[0]; a[13]=(_Float16)f3[1]; a[14]=(_Float16)f3[2]; a[15]=(_Float16)f3[3];

        // B: lane = expert column; lo lanes carry K=0..7,16..23,
        // hi lanes K=8..15,24..31; rows rotated by e*8 halfs.
        const int off  = d0 + lanehi * 8 + exp8 * 8;
        const v8h b0 = *(const v8h*)&Wh[exp8 * DDIM + ( off      & (DDIM - 1))];
        const v8h b1 = *(const v8h*)&Wh[exp8 * DDIM + ((off + 16) & (DDIM - 1))];
        v16h b;
        b[0]=b0[0]; b[1]=b0[1]; b[2]=b0[2]; b[3]=b0[3];
        b[4]=b0[4]; b[5]=b0[5]; b[6]=b0[6]; b[7]=b0[7];
        b[8]=b1[0]; b[9]=b1[1]; b[10]=b1[2]; b[11]=b1[3];
        b[12]=b1[4]; b[13]=b1[5]; b[14]=b1[6]; b[15]=b1[7];

        acc = __builtin_amdgcn_wmma_f32_16x16x32_f16(
            /*neg_a=*/false, a, /*neg_b=*/false, b,
            /*c_mod=*/(short)0, acc, /*reuse_a=*/false, /*reuse_b=*/false);
    }

    // ---- Phase 2: top-2 + softmax (logits via LDS overlay) ----
    __syncthreads();                      // everyone done reading Wh
    float* lg = (float*)smem;             // [TOK_PER_WG][NEXP] = 2 KB

    // C layout: VGPR r holds M=r (lanes 0-15) / M=r+8 (lanes 16-31), N=lane%16.
    if (lanelo < NEXP) {
        #pragma unroll
        for (int r = 0; r < 8; ++r)
            lg[((wave << 4) + r + (lanehi << 3)) * NEXP + lanelo] = acc[r];
    }
    __syncthreads();

    if (lane < 16) {
        const float* v = &lg[((wave << 4) + lane) * NEXP];
        // top-1 (ties -> lowest index, matching jax.lax.top_k)
        int i0 = 0; float v0 = v[0];
        #pragma unroll
        for (int j = 1; j < NEXP; ++j) { const float t = v[j]; if (t > v0) { v0 = t; i0 = j; } }
        // top-2
        int i1 = -1; float v1 = -3.4028235e38f;
        #pragma unroll
        for (int j = 0; j < NEXP; ++j) {
            if (j == i0) continue;
            const float t = v[j];
            if (t > v1) { v1 = t; i1 = j; }
        }
        const float e1 = __expf(v1 - v0);          // <= 1, no overflow
        const float w0 = 1.0f / (1.0f + e1);
        const int tok  = tok0 + lane;
        wout[tok * KSEL + 0] = w0;
        wout[tok * KSEL + 1] = e1 * w0;
        iout[tok * KSEL + 0] = i0;
        iout[tok * KSEL + 1] = i1;
    }
}

extern "C" void kernel_launch(void* const* d_in, const int* in_sizes, int n_in,
                              void* d_out, int out_size, void* d_ws, size_t ws_size,
                              hipStream_t stream) {
    (void)in_sizes; (void)n_in; (void)d_ws; (void)ws_size; (void)out_size;
    const float* x  = (const float*)d_in[0];
    const float* Wg = (const float*)d_in[1];
    // d_in[2] is top_k (==2), fixed at compile time.
    float* wout = (float*)d_out;                    // 32768 routing weights
    int*   iout = (int*)d_out + NTOK * KSEL;        // 32768 expert indices

    topk_router_wmma<<<NTOK / TOK_PER_WG, BLOCK, SMEM_BYTES, stream>>>(x, Wg, wout, iout);
}
